// MultiHeadAttention_63969242906730
// MI455X (gfx1250) — matrix operand order repaired and checked by reading, not compiled, not run on hardware
//
#include <hip/hip_runtime.h>
#include <math.h>

#define BB 2
#define NN 4096
#define DD 512
#define HH 8
#define HD 64
#define BN (BB*NN)

typedef __attribute__((ext_vector_type(8)))  __bf16 v8bf;
typedef __attribute__((ext_vector_type(16))) __bf16 v16bf;
typedef __attribute__((ext_vector_type(8)))  float  v8f;

static __device__ __forceinline__ __bf16 tobf(float f) {
    union { float f; unsigned int u; } a; a.f = f;
    unsigned int r = a.u + 0x7FFFu + ((a.u >> 16) & 1u);
    union { unsigned short u; __bf16 b; } o; o.u = (unsigned short)(r >> 16);
    return o.b;
}

// A-operand fragment (16-bit A 16x32): lane row = l&15, chunks at k = 8*hs and 16+8*hs
static __device__ __forceinline__ v16bf load_frag_A(const __bf16* rowptr, int hs) {
    union { v16bf v; v8bf h[2]; } u;
    u.h[0] = *(const v8bf*)(rowptr + 8*hs);
    u.h[1] = *(const v8bf*)(rowptr + 16 + 8*hs);
    return u.v;
}
// B-operand fragment (16-bit B 32x16): lane col = l&15, chunks at k = 8*hs and 16+8*hs
static __device__ __forceinline__ v16bf load_frag_B(const __bf16* rowptr, int hs) {
    union { v16bf v; v8bf h[2]; } u;
    u.h[0] = *(const v8bf*)(rowptr + 8*hs);
    u.h[1] = *(const v8bf*)(rowptr + 16 + 8*hs);
    return u.v;
}

static __device__ __forceinline__ v8f wmma_bf16(v16bf a, v16bf b, v8f c) {
    return __builtin_amdgcn_wmma_f32_16x16x32_bf16(false, a, false, b, (short)0, c, false, false);
}

// ---------------- conversion kernels ----------------
__global__ void f32_to_bf16_k(const float* __restrict__ src, __bf16* __restrict__ dst, int n) {
    int i = blockIdx.x * blockDim.x + threadIdx.x;
    if (i < n) dst[i] = tobf(src[i]);
}

// Wt[n][k] = bf16(W[k][n]), 512x512
__global__ void transpose_w_k(const float* __restrict__ W, __bf16* __restrict__ Wt) {
    int i = blockIdx.x * blockDim.x + threadIdx.x;
    int n = i >> 9, k = i & 511;
    Wt[i] = tobf(W[(size_t)k * DD + n]);
}

// ---------------- 8192x512x512 GEMM: C = X(bf16) @ Wt^T(bf16) + bias ----------------
// mode 0: Q  -> outb row-major, scaled          mode 1: K -> outb row-major
// mode 2: V  -> outb transposed [b][h][hd][N]   mode 3: O -> outf row-major fp32
__global__ void gemm512_k(const __bf16* __restrict__ X, const __bf16* __restrict__ Wt,
                          const float* __restrict__ bias, __bf16* __restrict__ outb,
                          float* __restrict__ outf, int mode, float scale) {
    const int n0 = blockIdx.x * 64, m0 = blockIdx.y * 64;
    const int t = threadIdx.x, w = t >> 5, l = t & 31, ln = l & 15, hs = l >> 4;
    __shared__ __align__(16) __bf16 As[64 * 64];
    __shared__ __align__(16) __bf16 Bs[64 * 64];

    v8f acc[4];
    const v8f z = {0.f,0.f,0.f,0.f,0.f,0.f,0.f,0.f};
    for (int s = 0; s < 4; ++s) acc[s] = z;

    // staging: 64x64 bf16 tile = 512 16B-chunks per array; 4 per thread.
    // chunk c = t + 128*j : row = srow + 16*j, byte-chunk sko
    const int srow = t >> 3;          // 0..15
    const int sko  = (t & 7) * 8;     // 0,8,...,56

    for (int k0 = 0; k0 < DD; k0 += 64) {
        v8bf a0 = *(const v8bf*)&X [(size_t)(m0 + srow     ) * DD + k0 + sko];
        v8bf a1 = *(const v8bf*)&X [(size_t)(m0 + srow + 16) * DD + k0 + sko];
        v8bf a2 = *(const v8bf*)&X [(size_t)(m0 + srow + 32) * DD + k0 + sko];
        v8bf a3 = *(const v8bf*)&X [(size_t)(m0 + srow + 48) * DD + k0 + sko];
        v8bf b0 = *(const v8bf*)&Wt[(size_t)(n0 + srow     ) * DD + k0 + sko];
        v8bf b1 = *(const v8bf*)&Wt[(size_t)(n0 + srow + 16) * DD + k0 + sko];
        v8bf b2 = *(const v8bf*)&Wt[(size_t)(n0 + srow + 32) * DD + k0 + sko];
        v8bf b3 = *(const v8bf*)&Wt[(size_t)(n0 + srow + 48) * DD + k0 + sko];
        if (k0 + 64 < DD) {   // pull next k-panel toward the WGP while we compute
            __builtin_prefetch(&X [(size_t)(m0 + srow) * DD + k0 + 64 + sko], 0, 3);
            __builtin_prefetch(&Wt[(size_t)(n0 + srow) * DD + k0 + 64 + sko], 0, 3);
        }
        *(v8bf*)&As[(srow     ) * 64 + sko] = a0;
        *(v8bf*)&As[(srow + 16) * 64 + sko] = a1;
        *(v8bf*)&As[(srow + 32) * 64 + sko] = a2;
        *(v8bf*)&As[(srow + 48) * 64 + sko] = a3;
        *(v8bf*)&Bs[(srow     ) * 64 + sko] = b0;
        *(v8bf*)&Bs[(srow + 16) * 64 + sko] = b1;
        *(v8bf*)&Bs[(srow + 32) * 64 + sko] = b2;
        *(v8bf*)&Bs[(srow + 48) * 64 + sko] = b3;
        __syncthreads();
#pragma unroll
        for (int kk = 0; kk < 64; kk += 32) {
            v16bf av = load_frag_A(&As[(16 * w + ln) * 64 + kk], hs);
#pragma unroll
            for (int s = 0; s < 4; ++s) {
                v16bf bv = load_frag_B(&Bs[(16 * s + ln) * 64 + kk], hs);
                acc[s] = wmma_bf16(av, bv, acc[s]);
            }
        }
        __syncthreads();
    }

#pragma unroll
    for (int s = 0; s < 4; ++s) {
        int n = n0 + 16 * s + ln;
        float bv = bias[n];
#pragma unroll
        for (int r = 0; r < 8; ++r) {
            int row = m0 + 16 * w + r + 8 * hs;       // global row in [0, BN)
            float v = acc[s][r] + bv;
            if (mode == 0) {
                outb[(size_t)row * DD + n] = tobf(v * scale);
            } else if (mode == 1) {
                outb[(size_t)row * DD + n] = tobf(v);
            } else if (mode == 2) {
                int b_ = row >> 12, m = row & 4095, h_ = n >> 6, dim = n & 63;
                outb[((size_t)(b_ * HH + h_) * HD + dim) * NN + m] = tobf(v);
            } else {
                outf[(size_t)row * DD + n] = v;
            }
        }
    }
}

// ---------------- pass 1: per-key-column max & sum(exp) over the QUERY axis ----------------
__global__ void colstats_k(const __bf16* __restrict__ Qb, const __bf16* __restrict__ Kb,
                           float* __restrict__ colmax, float* __restrict__ colsum) {
    const int bh = blockIdx.y, b = bh >> 3, h = bh & 7;
    const int m0 = blockIdx.x * 64;
    const int t = threadIdx.x, w = t >> 5, l = t & 31, ln = l & 15, hs = l >> 4;
    const v8f z = {0.f,0.f,0.f,0.f,0.f,0.f,0.f,0.f};

    v16bf kf0[4], kf1[4];
#pragma unroll
    for (int s = 0; s < 4; ++s) {
        const __bf16* Krow = Kb + ((size_t)(b * NN + m0 + 16 * s + ln) * DD + h * HD);
        kf0[s] = load_frag_B(Krow, hs);
        kf1[s] = load_frag_B(Krow + 32, hs);
    }

    float mx[4], sm[4];
#pragma unroll
    for (int s = 0; s < 4; ++s) { mx[s] = -3.0e38f; sm[s] = 0.f; }

    for (int n0 = 16 * w; n0 < NN; n0 += 64) {
        const __bf16* Qrow = Qb + ((size_t)(b * NN + n0 + ln) * DD + h * HD);
        if (n0 + 64 < NN)
            __builtin_prefetch(Qrow + (size_t)64 * DD, 0, 3);
        v16bf qa0 = load_frag_A(Qrow, hs);
        v16bf qa1 = load_frag_A(Qrow + 32, hs);
#pragma unroll
        for (int s = 0; s < 4; ++s) {
            v8f acc = z;
            acc = wmma_bf16(qa0, kf0[s], acc);
            acc = wmma_bf16(qa1, kf1[s], acc);
            float tmx = acc[0];
#pragma unroll
            for (int r = 1; r < 8; ++r) tmx = fmaxf(tmx, acc[r]);
            float tsm = 0.f;
#pragma unroll
            for (int r = 0; r < 8; ++r) tsm += __expf(acc[r] - tmx);
            float M = fmaxf(mx[s], tmx);
            sm[s] = sm[s] * __expf(mx[s] - M) + tsm * __expf(tmx - M);
            mx[s] = M;
        }
    }

    __shared__ float smx[4][64], ssm[4][64];
#pragma unroll
    for (int s = 0; s < 4; ++s) {
        float omx = __shfl_xor(mx[s], 16);
        float osm = __shfl_xor(sm[s], 16);
        float M = fmaxf(mx[s], omx);
        float S = sm[s] * __expf(mx[s] - M) + osm * __expf(omx - M);
        if (l < 16) { smx[w][16 * s + ln] = M; ssm[w][16 * s + ln] = S; }
    }
    __syncthreads();
    if (t < 64) {
        float M = smx[0][t];
        for (int wv = 1; wv < 4; ++wv) M = fmaxf(M, smx[wv][t]);
        float S = 0.f;
        for (int wv = 0; wv < 4; ++wv) S += ssm[wv][t] * __expf(smx[wv][t] - M);
        colmax[(size_t)bh * NN + m0 + t] = M;
        colsum[(size_t)bh * NN + m0 + t] = S;
    }
}

// ---------------- pass 2: recompute scores, apply column-softmax, P @ V ----------------
__global__ void attn_out_k(const __bf16* __restrict__ Qb, const __bf16* __restrict__ Kb,
                           const __bf16* __restrict__ Vt, const float* __restrict__ colmax,
                           const float* __restrict__ colsum, __bf16* __restrict__ attnb) {
    const int bh = blockIdx.y, b = bh >> 3, h = bh & 7;
    const int n0 = blockIdx.x * 64;
    const int t = threadIdx.x, w = t >> 5, l = t & 31, ln = l & 15, hs = l >> 4;
    const v8f z = {0.f,0.f,0.f,0.f,0.f,0.f,0.f,0.f};
    __shared__ __align__(16) __bf16 Pt[4][16][32];

    const __bf16* Qrow = Qb + ((size_t)(b * NN + n0 + 16 * w + ln) * DD + h * HD);
    v16bf qa0 = load_frag_A(Qrow, hs);
    v16bf qa1 = load_frag_A(Qrow + 32, hs);

    v8f oacc[4];
    for (int s = 0; s < 4; ++s) oacc[s] = z;

    const float* cmb = colmax + (size_t)bh * NN;
    const float* csb = colsum + (size_t)bh * NN;

    for (int m0 = 0; m0 < NN; m0 += 32) {
        if (m0 + 32 < NN) {   // stream next K rows / V panel ahead of the WMMAs
            __builtin_prefetch(Kb + ((size_t)(b * NN + m0 + 32 + ln) * DD + h * HD), 0, 3);
            __builtin_prefetch(Vt + ((size_t)(bh * HD + ln) * NN + m0 + 32), 0, 3);
        }
        v8f sacc[2];
#pragma unroll
        for (int sub = 0; sub < 2; ++sub) {
            sacc[sub] = z;
            const __bf16* Krow = Kb + ((size_t)(b * NN + m0 + 16 * sub + ln) * DD + h * HD);
            v16bf kb0 = load_frag_B(Krow, hs);
            v16bf kb1 = load_frag_B(Krow + 32, hs);
            sacc[sub] = wmma_bf16(qa0, kb0, sacc[sub]);
            sacc[sub] = wmma_bf16(qa1, kb1, sacc[sub]);
        }
        // p[n,m] = exp(s - colmax[m]) / colsum[m]; write to LDS transposed to A layout
#pragma unroll
        for (int sub = 0; sub < 2; ++sub) {
            int mcol = m0 + 16 * sub + ln;
            float cm = cmb[mcol];
            float rcs = 1.0f / csb[mcol];
#pragma unroll
            for (int r = 0; r < 8; ++r) {
                float p = __expf(sacc[sub][r] - cm) * rcs;
                Pt[w][r + 8 * hs][16 * sub + ln] = tobf(p);
            }
        }
        v16bf pa = load_frag_A(&Pt[w][ln][0], hs);
#pragma unroll
        for (int s = 0; s < 4; ++s) {
            const __bf16* Vrow = Vt + ((size_t)(bh * HD + 16 * s + ln) * NN + m0);
            v16bf vb = load_frag_B(Vrow, hs);
            oacc[s] = wmma_bf16(pa, vb, oacc[s]);
        }
    }

#pragma unroll
    for (int s = 0; s < 4; ++s)
#pragma unroll
        for (int r = 0; r < 8; ++r) {
            int n = n0 + 16 * w + r + 8 * hs;
            attnb[(size_t)(b * NN + n) * DD + h * HD + 16 * s + ln] = tobf(oacc[s][r]);
        }
}

// ---------------- host ----------------
extern "C" void kernel_launch(void* const* d_in, const int* in_sizes, int n_in,
                              void* d_out, int out_size, void* d_ws, size_t ws_size,
                              hipStream_t stream) {
    (void)in_sizes; (void)n_in; (void)out_size; (void)ws_size;
    const float* x_q = (const float*)d_in[0];
    const float* x_k = (const float*)d_in[1];
    const float* x_v = (const float*)d_in[2];
    // d_in[3] = mask: dead code in the reference
    const float* W_q = (const float*)d_in[4];
    const float* b_q = (const float*)d_in[5];
    const float* W_k = (const float*)d_in[6];
    const float* b_k = (const float*)d_in[7];
    const float* W_v = (const float*)d_in[8];
    const float* b_v = (const float*)d_in[9];
    const float* W_o = (const float*)d_in[10];
    const float* b_o = (const float*)d_in[11];

    char* p = (char*)d_ws;
    auto carve = [&](size_t bytes) { char* r = p; p += (bytes + 255) & ~(size_t)255; return (void*)r; };
    const size_t xb = (size_t)BN * DD * sizeof(__bf16);
    const size_t wb = (size_t)DD * DD * sizeof(__bf16);
    __bf16* Xqb = (__bf16*)carve(xb);
    __bf16* Xkb = (__bf16*)carve(xb);
    __bf16* Xvb = (__bf16*)carve(xb);
    __bf16* Wqt = (__bf16*)carve(wb);
    __bf16* Wkt = (__bf16*)carve(wb);
    __bf16* Wvt = (__bf16*)carve(wb);
    __bf16* Wot = (__bf16*)carve(wb);
    __bf16* Qb  = (__bf16*)carve(xb);
    __bf16* Kb  = (__bf16*)carve(xb);
    __bf16* Vt  = (__bf16*)carve(xb);
    __bf16* Attnb = (__bf16*)carve(xb);
    float* colmax = (float*)carve((size_t)BB * HH * NN * sizeof(float));
    float* colsum = (float*)carve((size_t)BB * HH * NN * sizeof(float));

    const int xcnt = BN * DD;
    f32_to_bf16_k<<<(xcnt + 255) / 256, 256, 0, stream>>>(x_q, Xqb, xcnt);
    f32_to_bf16_k<<<(xcnt + 255) / 256, 256, 0, stream>>>(x_k, Xkb, xcnt);
    f32_to_bf16_k<<<(xcnt + 255) / 256, 256, 0, stream>>>(x_v, Xvb, xcnt);
    transpose_w_k<<<(DD * DD) / 256, 256, 0, stream>>>(W_q, Wqt);
    transpose_w_k<<<(DD * DD) / 256, 256, 0, stream>>>(W_k, Wkt);
    transpose_w_k<<<(DD * DD) / 256, 256, 0, stream>>>(W_v, Wvt);
    transpose_w_k<<<(DD * DD) / 256, 256, 0, stream>>>(W_o, Wot);

    const float scale = 1.0f / sqrtf((float)DD);   // quirk: sqrt(hidden), folded into Q
    dim3 gg(DD / 64, BN / 64), bb(128);
    gemm512_k<<<gg, bb, 0, stream>>>(Xqb, Wqt, b_q, Qb, nullptr, 0, scale);
    gemm512_k<<<gg, bb, 0, stream>>>(Xkb, Wkt, b_k, Kb, nullptr, 1, 1.0f);
    gemm512_k<<<gg, bb, 0, stream>>>(Xvb, Wvt, b_v, Vt, nullptr, 2, 1.0f);

    colstats_k<<<dim3(NN / 64, BB * HH), 128, 0, stream>>>(Qb, Kb, colmax, colsum);
    attn_out_k<<<dim3(NN / 64, BB * HH), 128, 0, stream>>>(Qb, Kb, Vt, colmax, colsum, Attnb);

    gemm512_k<<<gg, bb, 0, stream>>>(Attnb, Wot, b_o, nullptr, (float*)d_out, 3, 1.0f);
}
